// SpectralConv2d_RC_51376398795189
// MI455X (gfx1250) — matrix operand order repaired and checked by reading, not compiled, not run on hardware
//
#include <hip/hip_runtime.h>
#include <math.h>

// ---------------------------------------------------------------------------
// Types / WMMA helpers (CDNA5 gfx1250, wave32)
// ---------------------------------------------------------------------------
typedef _Float16 half_t;
typedef __attribute__((ext_vector_type(16))) _Float16 v16h;
typedef __attribute__((ext_vector_type(8)))  float    v8f;

#define WMMA_F16(A, B, C) \
  __builtin_amdgcn_wmma_f32_16x16x32_f16(false, (A), false, (B), (short)0, (C), false, false)

// A-fragment (M=16 x K=32, f16), matrix stored row-major with 'stride' halfs
// per row; 'base' points at (row0, k0). Layout per ISA 7.12.2:
//   lane L: m = L%16, g = L/16;  half i<8 -> k = g*8+i ;  i>=8 -> k = 16+g*8+(i-8)
__device__ __forceinline__ v16h ldA(const half_t* base, int stride, int lane) {
  const int m = lane & 15, g = lane >> 4;
  const half_t* p = base + m * stride + g * 8;
  v16h r;
#pragma unroll
  for (int i = 0; i < 8; ++i) { r[i] = p[i]; r[i + 8] = p[i + 16]; }
  return r;
}

// A-fragment with logical K=16 zero-padded to 32 (rows hold only 16 halfs).
__device__ __forceinline__ v16h ldA_pad16(const half_t* base, int stride, int lane) {
  const int m = lane & 15, g = lane >> 4;
  const half_t* p = base + m * stride + g * 8;
  v16h r;
#pragma unroll
  for (int i = 0; i < 8; ++i) { r[i] = p[i]; r[i + 8] = (half_t)0.0f; }
  return r;
}

// B-fragment (K=32 x N=16, f16). Matrix stored TRANSPOSED [N][K] with 'stride'
// halfs per N-row so the 16 K-values each lane needs are contiguous:
//   lane L: n = L%16, g = L/16; half i<8 -> k = g*8+i ;  i>=8 -> k = 16+g*8+(i-8)
__device__ __forceinline__ v16h ldB(const half_t* base, int stride, int lane) {
  const int n = lane & 15, g = lane >> 4;
  const half_t* p = base + n * stride + g * 8;
  v16h r;
#pragma unroll
  for (int i = 0; i < 8; ++i) { r[i] = p[i]; r[i + 8] = p[i + 16]; }
  return r;
}

__device__ __forceinline__ v16h vneg(v16h a) {
  v16h r;
#pragma unroll
  for (int i = 0; i < 16; ++i) r[i] = -a[i];
  return r;
}

// ---------------------------------------------------------------------------
// Problem constants
// ---------------------------------------------------------------------------
#define NB   32
#define NC   32
#define NH   128
#define NW   128
#define KH   32   // retained H modes: rows 0..15 (weights1) and 112..127 (weights2)
#define KW   16
#define ANG_STEP 0.049087385212340519350978802863742f   // 2*pi/128

// ---------------------------------------------------------------------------
// Kernel E1: DFT twiddle matrices, f16, WMMA-fragment-friendly layouts.
//   fwt  : forward W-DFT, B-operand, transposed [kw(16)][w(128)], re then im(-sin)
//   fh   : forward H-DFT, A-operand, [kh(32)][h(128)], re then im(-sin)
//   gwt  : inverse W (irfft col factors * s_kw/128), B-operand transposed
//          [w(128)][kwpad(32)] (kw>=16 zero), re then im(+sin)
//   gh   : inverse H (1/128 * e^{+i th}), A-operand [h(128)][kh(32)],
//          re then NEGATED im (so y = GhRe@Sre + GhImN@Sim)
// ---------------------------------------------------------------------------
__global__ __launch_bounds__(256) void twiddle_kernel(half_t* fwt, half_t* fh,
                                                      half_t* gwt, half_t* gh) {
  const int gid = blockIdx.x * blockDim.x + threadIdx.x;
  const int stride = gridDim.x * blockDim.x;
  for (int i = gid; i < 16 * 128; i += stride) {        // fwt [kw][w]
    int kw = i >> 7, w = i & 127;
    float a = (float)((kw * w) & 127) * ANG_STEP;
    fwt[i] = (half_t)cosf(a);
    fwt[2048 + i] = (half_t)(-sinf(a));
  }
  for (int i = gid; i < 32 * 128; i += stride) {        // fh [kh][h]
    int kh = i >> 7, h = i & 127;
    int r = (kh < 16) ? kh : (96 + kh);
    float a = (float)((r * h) & 127) * ANG_STEP;
    fh[i] = (half_t)cosf(a);
    fh[4096 + i] = (half_t)(-sinf(a));
  }
  for (int i = gid; i < 128 * 32; i += stride) {        // gwt [w][kwpad]
    int w = i >> 5, kp = i & 31;
    float re = 0.f, im = 0.f;
    if (kp < 16) {
      float s = (kp == 0) ? (1.0f / 128.0f) : (2.0f / 128.0f);
      float a = (float)((kp * w) & 127) * ANG_STEP;
      re = s * cosf(a); im = s * sinf(a);
    }
    gwt[i] = (half_t)re;
    gwt[4096 + i] = (half_t)im;
  }
  for (int i = gid; i < 128 * 32; i += stride) {        // gh [h][kh]
    int h = i >> 5, kh = i & 31;
    int r = (kh < 16) ? kh : (96 + kh);
    float a = (float)((r * h) & 127) * ANG_STEP;
    gh[i] = (half_t)(cosf(a) * (1.0f / 128.0f));
    gh[4096 + i] = (half_t)(-sinf(a) * (1.0f / 128.0f));   // negated imag
  }
}

// ---------------------------------------------------------------------------
// Kernel E2: pack spectral weights as per-mode f16 B-operands [mode][comp][o][i]
// ---------------------------------------------------------------------------
__global__ __launch_bounds__(256) void pack_weights_kernel(const float* w1r, const float* w1i,
                                                           const float* w2r, const float* w2i,
                                                           half_t* wpack) {
  const int mode = blockIdx.x;            // kh*16+kw
  const int kh = mode >> 4, kw = mode & 15;
  for (int e = 0; e < 4; ++e) {
    int id = threadIdx.x + e * 256;       // 0..1023
    int o = id >> 5, i = id & 31;
    float wr, wi;
    if (kh < 16) {
      int s = ((i * 32 + o) * 16 + kh) * 16 + kw;
      wr = w1r[s]; wi = w1i[s];
    } else {
      int s = ((i * 32 + o) * 16 + (kh - 16)) * 16 + kw;
      wr = w2r[s]; wi = w2i[s];
    }
    size_t base = (size_t)mode * 2048;
    wpack[base + (size_t)o * 32 + i] = (half_t)wr;
    wpack[base + 1024 + (size_t)o * 32 + i] = (half_t)wi;
  }
}

// ---------------------------------------------------------------------------
// Stage A: truncated forward 2-D DFT.  One workgroup per (b,c), 4 waves.
//   T1[h,kw] = x[h,:] @ Fw     (M=128,N=16,K=128) re/im
//   XF[kh,kw] = Fh @ T1        (M=32, N=16, K=128) complex
//   XF stored f16 mode-major: [mode(512)][comp(2)][b(32)][i=c(32)]
// ---------------------------------------------------------------------------
__global__ __launch_bounds__(128) void fwd_kernel(const float* __restrict__ x,
                                                  const half_t* __restrict__ fwt,
                                                  const half_t* __restrict__ fh,
                                                  half_t* __restrict__ xfws) {
  __shared__ half_t xh[128 * 128];            // x tile, f16
  __shared__ half_t t1t[2 * 16 * 128];        // T1 transposed [comp][kw][h]
  const int bc = blockIdx.x;
  const int b = bc >> 5, c = bc & 31;
  const int tid = threadIdx.x, lane = tid & 31, wave = tid >> 5;

  // load + convert one row per thread
  {
    const float* row = x + (size_t)bc * (128 * 128) + (size_t)tid * 128;
    __builtin_prefetch(row, 0, 1);
#pragma unroll 4
    for (int w4 = 0; w4 < 32; ++w4) {
      float4 v = ((const float4*)row)[w4];
      int o = tid * 128 + w4 * 4;
      xh[o + 0] = (half_t)v.x; xh[o + 1] = (half_t)v.y;
      xh[o + 2] = (half_t)v.z; xh[o + 3] = (half_t)v.w;
    }
  }
  __syncthreads();

  // Step 1: T1 (16 tile-jobs = 8 M-tiles x 2 components over 4 waves)
  for (int jj = 0; jj < 4; ++jj) {
    const int job = jj * 4 + wave;          // 0..15
    const int comp = job >> 3, mt = job & 7;
    const half_t* bmat = fwt + comp * 2048; // [16][128] transposed B
    v8f acc = {};
#pragma unroll
    for (int ks = 0; ks < 4; ++ks) {
      v16h a = ldA(xh + mt * 16 * 128 + ks * 32, 128, lane);
      v16h bf = ldB(bmat + ks * 32, 128, lane);
      acc = WMMA_F16(a, bf, acc);
    }
    // store transposed into t1t[comp][kw][h]
    half_t* dst = t1t + comp * 2048 + (lane & 15) * 128 + mt * 16 + (lane >> 4) * 8;
#pragma unroll
    for (int r = 0; r < 8; ++r) dst[r] = (half_t)acc[r];
  }
  __syncthreads();

  // Step 2: XF (4 jobs: comp x 2 kh-tiles, one per wave)
  {
    const int comp = wave >> 1, mt = wave & 1;
    v8f acc = {};
#pragma unroll
    for (int ks = 0; ks < 4; ++ks) {
      v16h ar = ldA(fh + mt * 16 * 128 + ks * 32, 128, lane);          // FhRe
      v16h ai = ldA(fh + 4096 + mt * 16 * 128 + ks * 32, 128, lane);   // FhIm
      v16h br = ldB(t1t + ks * 32, 128, lane);                         // T1re
      v16h bi = ldB(t1t + 2048 + ks * 32, 128, lane);                  // T1im
      if (comp == 0) {   // XFre = FhRe@T1re - FhIm@T1im
        acc = WMMA_F16(ar, br, acc);
        acc = WMMA_F16(vneg(ai), bi, acc);
      } else {           // XFim = FhRe@T1im + FhIm@T1re
        acc = WMMA_F16(ar, bi, acc);
        acc = WMMA_F16(ai, br, acc);
      }
    }
#pragma unroll
    for (int r = 0; r < 8; ++r) {
      int kh = mt * 16 + (lane >> 4) * 8 + r;
      int kw = lane & 15;
      size_t idx = ((((size_t)(kh * 16 + kw) * 2 + comp) * 32 + b) * 32 + c);
      xfws[idx] = (half_t)acc[r];
    }
  }
}

// ---------------------------------------------------------------------------
// Stage B: per-mode channel mixing  Z[b,o] = sum_i XF[b,i] * W[i,o]  (complex)
// One workgroup per mode (512), 4 waves = 4 output quadrants, K=32 = 1 WMMA.
// Z stored f16: [b][o][comp][kh(32)][kw(16)]
// ---------------------------------------------------------------------------
__global__ __launch_bounds__(128) void mix_kernel(const half_t* __restrict__ xfws,
                                                  const half_t* __restrict__ wpack,
                                                  half_t* __restrict__ zws) {
  const int mode = blockIdx.x;
  const int kh = mode >> 4, kw = mode & 15;
  const int tid = threadIdx.x, lane = tid & 31, wave = tid >> 5;
  const int bt = wave >> 1, ot = wave & 1;

  const half_t* xb = xfws + (size_t)mode * 2048;    // [comp][b][i]
  const half_t* wb = wpack + (size_t)mode * 2048;   // [comp][o][i] (B transposed)
  v16h are = ldA(xb + bt * 16 * 32, 32, lane);
  v16h aim = ldA(xb + 1024 + bt * 16 * 32, 32, lane);
  v16h wre = ldB(wb + ot * 16 * 32, 32, lane);
  v16h wim = ldB(wb + 1024 + ot * 16 * 32, 32, lane);

  v8f zre = {};
  zre = WMMA_F16(vneg(aim), wim, zre);
  zre = WMMA_F16(are, wre, zre);
  v8f zim = {};
  zim = WMMA_F16(aim, wre, zim);
  zim = WMMA_F16(are, wim, zim);

#pragma unroll
  for (int r = 0; r < 8; ++r) {
    int bb = bt * 16 + (lane >> 4) * 8 + r;
    int oo = ot * 16 + (lane & 15);
    size_t base = (((size_t)(bb * 32 + oo) * 2) * 32 + kh) * 16 + kw;
    zws[base] = (half_t)zre[r];
    zws[base + 512] = (half_t)zim[r];      // comp stride = 32*16
  }
}

// ---------------------------------------------------------------------------
// Stage C: inverse transform + corr add.  One workgroup per (b,o), 8 waves.
//   S[kh,w] = Z @ Gw    (M=32,N=128,K=16 padded to 32) complex, kept in LDS^T
//   y[h,w]  = GhRe@Sre + GhImN@Sim   (M=128,N=128,K=32)  -> out + corr[b,o]
// ---------------------------------------------------------------------------
__global__ __launch_bounds__(256) void inv_kernel(const half_t* __restrict__ zws,
                                                  const half_t* __restrict__ gwt,
                                                  const half_t* __restrict__ gh,
                                                  const float* __restrict__ corr,
                                                  float* __restrict__ out) {
  __shared__ half_t st[2 * 128 * 32];   // S transposed [comp][w][kh]
  const int bo = blockIdx.x;
  const int b = bo >> 5, o = bo & 31;
  const int tid = threadIdx.x, lane = tid & 31, wave = tid >> 5;
  const half_t* zb = zws + (size_t)bo * 1024;   // [comp][kh(32)][kw(16)]

  // Step 1: 32 jobs (mt2 x nt8 x comp2) over 8 waves
  for (int jj = 0; jj < 4; ++jj) {
    const int job = wave + 8 * jj;       // 0..31
    const int comp = job & 1, nt = (job >> 1) & 7, mt = job >> 4;
    v16h azre = ldA_pad16(zb + mt * 16 * 16, 16, lane);
    v16h azim = ldA_pad16(zb + 512 + mt * 16 * 16, 16, lane);
    v16h bre = ldB(gwt + nt * 16 * 32, 32, lane);
    v16h bim = ldB(gwt + 4096 + nt * 16 * 32, 32, lane);
    v8f acc = {};
    if (comp == 0) {   // Sre = Zre@GwRe - Zim@GwIm
      acc = WMMA_F16(vneg(azim), bim, acc);
      acc = WMMA_F16(azre, bre, acc);
    } else {           // Sim = Zre@GwIm + Zim@GwRe
      acc = WMMA_F16(azim, bre, acc);
      acc = WMMA_F16(azre, bim, acc);
    }
    half_t* dst = st + comp * 4096 + (nt * 16 + (lane & 15)) * 32 + mt * 16 + (lane >> 4) * 8;
#pragma unroll
    for (int r = 0; r < 8; ++r) dst[r] = (half_t)acc[r];
  }
  __syncthreads();

  // Step 2: 64 output tiles over 8 waves
  const float cv = corr[b * 32 + o];
  float* outp = out + (size_t)bo * (128 * 128);
  for (int jj = 0; jj < 8; ++jj) {
    const int tile = wave * 8 + jj;      // 0..63
    const int ht = tile >> 3, wt = tile & 7;
    v16h a0 = ldA(gh + ht * 16 * 32, 32, lane);          // GhRe
    v16h a1 = ldA(gh + 4096 + ht * 16 * 32, 32, lane);   // -GhIm (pre-negated)
    v16h b0 = ldB(st + wt * 16 * 32, 32, lane);          // Sre^T
    v16h b1 = ldB(st + 4096 + wt * 16 * 32, 32, lane);   // Sim^T
    v8f acc = {};
    acc = WMMA_F16(a0, b0, acc);
    acc = WMMA_F16(a1, b1, acc);
#pragma unroll
    for (int r = 0; r < 8; ++r) {
      int h = ht * 16 + (lane >> 4) * 8 + r;
      int w = wt * 16 + (lane & 15);
      outp[(size_t)h * 128 + w] = acc[r] + cv;
    }
  }
}

// ---------------------------------------------------------------------------
// CFT path — setup: A matrices [T=128, F=4] complex via piecewise Chebyshev
// least squares (normal-equations pinv, 8x8 Gauss-Jordan) + 16-pt Gauss-
// Legendre quadrature.  awah = Aw[128][4][2] then Ah[128][4][2] (f32).
// ---------------------------------------------------------------------------
__device__ const float GLX[16] = {
  -0.98940093499164993f, -0.94457502307323258f, -0.86563120238783174f, -0.75540440835500303f,
  -0.61787624440264375f, -0.45801677765722739f, -0.28160355077925891f, -0.09501250983763744f,
   0.09501250983763744f,  0.28160355077925891f,  0.45801677765722739f,  0.61787624440264375f,
   0.75540440835500303f,  0.86563120238783174f,  0.94457502307323258f,  0.98940093499164993f };
__device__ const float GLW[16] = {
   0.02715245941175409f, 0.06225352393864789f, 0.09515851168249278f, 0.12462897125553387f,
   0.14959598881657673f, 0.16915651939500254f, 0.18260341504492359f, 0.18945061045506850f,
   0.18945061045506850f, 0.18260341504492359f, 0.16915651939500254f, 0.14959598881657673f,
   0.12462897125553387f, 0.09515851168249278f, 0.06225352393864789f, 0.02715245941175409f };

__global__ __launch_bounds__(128) void cft_setup_kernel(float* awah) {
  __shared__ float V[128][8];
  __shared__ float M8r[8][4], M8i[8][4];
  const int tid = threadIdx.x;
  for (int target = 0; target < 2; ++target) {
    float* A = awah + target * 1024;
    float fv[4];
    if (target == 0) { fv[0] = 0.f; fv[1] = 1.f; fv[2] = 2.f; fv[3] = 3.f; }
    else             { fv[0] = 0.f; fv[1] = 1.f; fv[2] = -2.f; fv[3] = -1.f; }
    for (int i = tid; i < 1024; i += 128) A[i] = 0.f;
    __syncthreads();
    for (int l = 0; l < 4; ++l) {
      {
        float t = (float)tid / 127.0f;
        int seg = (int)floorf(t * 4.0f);
        seg = seg < 0 ? 0 : (seg > 3 ? 3 : seg);
        float cc = 0.25f * l + 0.125f;
        float u = (t - cc) * 8.0f;
        float mask = (seg == l) ? 1.f : 0.f;
        float tm2 = 1.f, tm1 = u;
        V[tid][0] = mask; V[tid][1] = mask * u;
        for (int m = 2; m < 8; ++m) { float tm = 2.f * u * tm1 - tm2; V[tid][m] = mask * tm; tm2 = tm1; tm1 = tm; }
      }
      __syncthreads();
      if (tid == 0) {
        float G[8][16];
        for (int m = 0; m < 8; ++m)
          for (int k = 0; k < 16; ++k) G[m][k] = (k >= 8) ? ((k - 8 == m) ? 1.f : 0.f) : 0.f;
        for (int i = 0; i < 128; ++i)
          for (int m = 0; m < 8; ++m)
            for (int mp = 0; mp < 8; ++mp) G[m][mp] += V[i][m] * V[i][mp];
        // Gauss-Jordan with partial pivoting -> G[:][8..15] = inverse
        for (int col = 0; col < 8; ++col) {
          int piv = col; float best = fabsf(G[col][col]);
          for (int r2 = col + 1; r2 < 8; ++r2) { float v = fabsf(G[r2][col]); if (v > best) { best = v; piv = r2; } }
          if (piv != col) for (int k = 0; k < 16; ++k) { float tmp = G[col][k]; G[col][k] = G[piv][k]; G[piv][k] = tmp; }
          float inv = 1.0f / G[col][col];
          for (int k = 0; k < 16; ++k) G[col][k] *= inv;
          for (int r2 = 0; r2 < 8; ++r2) if (r2 != col) {
            float f = G[r2][col];
            for (int k = 0; k < 16; ++k) G[r2][k] -= f * G[col][k];
          }
        }
        // quadrature integrals I[m][f] (complex)
        float Ir[8][4], Ii[8][4];
        for (int m = 0; m < 8; ++m) for (int f = 0; f < 4; ++f) { Ir[m][f] = 0.f; Ii[m][f] = 0.f; }
        float cc = 0.25f * l + 0.125f;
        for (int q = 0; q < 16; ++q) {
          float uq = GLX[q], wq = GLW[q];
          float Tq[8]; Tq[0] = 1.f; Tq[1] = uq;
          for (int m = 2; m < 8; ++m) Tq[m] = 2.f * uq * Tq[m - 1] - Tq[m - 2];
          float tq = 0.125f * uq + cc;
          for (int f = 0; f < 4; ++f) {
            float th = 6.2831853071795864769f * tq * fv[f];
            float cs = cosf(th), sn = sinf(th);
            for (int m = 0; m < 8; ++m) {
              Ir[m][f] += 0.125f * wq * Tq[m] * cs;
              Ii[m][f] -= 0.125f * wq * Tq[m] * sn;
            }
          }
        }
        // M8 = Ginv @ I
        for (int m = 0; m < 8; ++m)
          for (int f = 0; f < 4; ++f) {
            float sr = 0.f, si = 0.f;
            for (int mp = 0; mp < 8; ++mp) { sr += G[m][8 + mp] * Ir[mp][f]; si += G[m][8 + mp] * Ii[mp][f]; }
            M8r[m][f] = sr; M8i[m][f] = si;
          }
      }
      __syncthreads();
      for (int f = 0; f < 4; ++f) {
        float sr = 0.f, si = 0.f;
        for (int m = 0; m < 8; ++m) { sr += V[tid][m] * M8r[m][f]; si += V[tid][m] * M8i[m][f]; }
        A[(tid * 4 + f) * 2 + 0] += sr;
        A[(tid * 4 + f) * 2 + 1] += si;
      }
      __syncthreads();
    }
    __syncthreads();
  }
}

// cft[b,c,m,f] = sum_h Ah[h,m] * ( sum_w x[h,w] * Aw[w,f] )   (complex)
__global__ __launch_bounds__(128) void cft_kernel(const float* __restrict__ x,
                                                  const float* __restrict__ awah,
                                                  float* __restrict__ cftf) {
  __shared__ float t2[128][8];   // [h][f*2+comp]
  const int bc = blockIdx.x;
  const int b = bc >> 5, c = bc & 31;
  const int tid = threadIdx.x;
  const float* aw = awah;
  const float* ah = awah + 1024;
  {
    const float* row = x + (size_t)bc * (128 * 128) + (size_t)tid * 128;
    float acc[8];
#pragma unroll
    for (int k = 0; k < 8; ++k) acc[k] = 0.f;
    for (int w = 0; w < 128; ++w) {
      float xv = row[w];
#pragma unroll
      for (int f = 0; f < 4; ++f) {
        acc[f * 2 + 0] += xv * aw[(w * 4 + f) * 2 + 0];
        acc[f * 2 + 1] += xv * aw[(w * 4 + f) * 2 + 1];
      }
    }
#pragma unroll
    for (int k = 0; k < 8; ++k) t2[tid][k] = acc[k];
  }
  __syncthreads();
  if (tid < 32) {
    const int m = tid >> 3, f = (tid >> 1) & 3, comp = tid & 1;
    float s = 0.f;
    for (int h = 0; h < 128; ++h) {
      float ar = ah[(h * 4 + m) * 2 + 0], ai = ah[(h * 4 + m) * 2 + 1];
      float tr = t2[h][f * 2 + 0], ti = t2[h][f * 2 + 1];
      s += (comp == 0) ? (ar * tr - ai * ti) : (ar * ti + ai * tr);
    }
    cftf[(size_t)b * 1024 + ((c * 4 + m) * 4 + f) * 2 + comp] = s;
  }
}

// MLP: hmid = gelu(cft @ gw1^T + gb1);  corr = hmid @ gw2^T + gb2
__global__ __launch_bounds__(128) void mlp_kernel(const float* __restrict__ cftf,
                                                  const float* __restrict__ gw1,
                                                  const float* __restrict__ gb1,
                                                  const float* __restrict__ gw2,
                                                  const float* __restrict__ gb2,
                                                  float* __restrict__ corr) {
  __shared__ float cf[1024];
  __shared__ float hm[128];
  const int b = blockIdx.x, tid = threadIdx.x;
  for (int k = tid; k < 1024; k += 128) cf[k] = cftf[(size_t)b * 1024 + k];
  __syncthreads();
  {
    const float* wrow = gw1 + (size_t)tid * 1024;
    float s = gb1[tid];
    for (int k = 0; k < 1024; ++k) s += cf[k] * wrow[k];
    hm[tid] = 0.5f * s * (1.0f + erff(s * 0.70710678118654752f));   // exact GELU
  }
  __syncthreads();
  if (tid < 32) {
    const float* wrow = gw2 + (size_t)tid * 128;
    float s = gb2[tid];
    for (int j = 0; j < 128; ++j) s += hm[j] * wrow[j];
    corr[b * 32 + tid] = s;
  }
}

// ---------------------------------------------------------------------------
// Launch
// ---------------------------------------------------------------------------
extern "C" void kernel_launch(void* const* d_in, const int* in_sizes, int n_in,
                              void* d_out, int out_size, void* d_ws, size_t ws_size,
                              hipStream_t stream) {
  (void)in_sizes; (void)n_in; (void)out_size; (void)ws_size;
  const float* x   = (const float*)d_in[0];
  const float* w1r = (const float*)d_in[1];
  const float* w1i = (const float*)d_in[2];
  const float* w2r = (const float*)d_in[3];
  const float* w2i = (const float*)d_in[4];
  const float* gw1 = (const float*)d_in[5];
  const float* gb1 = (const float*)d_in[6];
  const float* gw2 = (const float*)d_in[7];
  const float* gb2 = (const float*)d_in[8];
  float* out = (float*)d_out;

  char* p = (char*)d_ws;
  auto take = [&](size_t n) { char* r = p; p += (n + 255) & ~(size_t)255; return r; };
  half_t* fwt   = (half_t*)take(4096ull * sizeof(half_t));
  half_t* fh    = (half_t*)take(8192ull * sizeof(half_t));
  half_t* gwt   = (half_t*)take(8192ull * sizeof(half_t));
  half_t* gh    = (half_t*)take(8192ull * sizeof(half_t));
  half_t* wpack = (half_t*)take(1048576ull * sizeof(half_t));
  half_t* xfws  = (half_t*)take(1048576ull * sizeof(half_t));
  half_t* zws   = (half_t*)take(1048576ull * sizeof(half_t));
  float*  awah  = (float*)take(2048ull * sizeof(float));
  float*  cftf  = (float*)take(32768ull * sizeof(float));
  float*  corr  = (float*)take(1024ull * sizeof(float));

  twiddle_kernel<<<32, 256, 0, stream>>>(fwt, fh, gwt, gh);
  pack_weights_kernel<<<512, 256, 0, stream>>>(w1r, w1i, w2r, w2i, wpack);
  fwd_kernel<<<1024, 128, 0, stream>>>(x, fwt, fh, xfws);
  mix_kernel<<<512, 128, 0, stream>>>(xfws, wpack, zws);
  cft_setup_kernel<<<1, 128, 0, stream>>>(awah);
  cft_kernel<<<1024, 128, 0, stream>>>(x, awah, cftf);
  mlp_kernel<<<32, 128, 0, stream>>>(cftf, gw1, gb1, gw2, gb2, corr);
  inv_kernel<<<1024, 256, 0, stream>>>(zws, gwt, gh, corr, out);
}
